// SparseHyperbolicAttention_79001628442773
// MI455X (gfx1250) — matrix operand order repaired and checked by reading, not compiled, not run on hardware
//
#include <hip/hip_runtime.h>

#define BATCH 4
#define SEQ   4096
#define DM    256
#define BS    64
#define NB    64
#define TOPK  4
#define SCALE 0.0625f   // 1/sqrt(256)

typedef __attribute__((ext_vector_type(16))) _Float16 v16h;
typedef __attribute__((ext_vector_type(8)))  float    v8f;

// LDS strides (elements), padded to avoid 64-bank conflicts
#define QS_S 260
#define KS_S 260
#define VT_S 66
#define SC_S 65
#define PS_S 68

// ---------------------------------------------------------------------------
// A-fragment (16x32 f16), contiguous k order (matches frag_b16):
// lanes 0-15: row m=lane, V0..7 -> K=0..15
// lanes 16-31: row m=lane-16, V0..7 -> K=16..31
__device__ __forceinline__ v16h frag_a16(const _Float16* base, int stride,
                                         int lane, int k0) {
  const int m    = lane & 15;
  const int koff = (lane >> 4) << 4;
  const _Float16* row = base + m * stride;
  v16h a;
#pragma unroll
  for (int g = 0; g < 8; ++g) {
    const int kk = k0 + koff + (g << 1);
    a[2 * g]     = row[kk];
    a[2 * g + 1] = row[kk + 1];
  }
  return a;
}

// B-fragment (32x16 f16): lane n holds column n; lanes 0-15 K=0..15 in V0..7,
// lanes 16-31 K=16..31.  base is indexed base[n*stride + k].
__device__ __forceinline__ v16h frag_b16(const _Float16* base, int stride,
                                         int lane, int k0) {
  const int n  = lane & 15;
  const int kb = k0 + ((lane >> 4) << 4);
  const _Float16* row = base + n * stride;
  v16h b;
#pragma unroll
  for (int g = 0; g < 8; ++g) {
    b[2 * g]     = row[kb + 2 * g];
    b[2 * g + 1] = row[kb + 2 * g + 1];
  }
  return b;
}

// ---------------------------------------------------------------------------
// Kernel 1: block centroids of q and k.  grid (NB, BATCH, 2), 256 threads.
__global__ void bsa_centroid_kernel(const float* __restrict__ q,
                                    const float* __restrict__ k,
                                    float* __restrict__ qc,
                                    float* __restrict__ kc) {
  const int blk = blockIdx.x, b = blockIdx.y;
  const float* src = (blockIdx.z == 0) ? q : k;
  float* dst       = (blockIdx.z == 0) ? qc : kc;
  const int d = threadIdx.x;
  const size_t base = ((size_t)b * SEQ + (size_t)blk * BS) * DM + d;
  float s = 0.f;
#pragma unroll 8
  for (int r = 0; r < BS; ++r) s += src[base + (size_t)r * DM];
  dst[((size_t)b * NB + blk) * DM + d] = s * (1.0f / BS);
}

// ---------------------------------------------------------------------------
// Kernel 2: top-4 nearest k-block centroids per q-block.  grid (NB, BATCH), 64 thr.
__global__ void bsa_topk_kernel(const float* __restrict__ qc,
                                const float* __restrict__ kc,
                                int* __restrict__ bidx) {
  __shared__ float dist[NB];
  const int qb = blockIdx.x, b = blockIdx.y;
  const int j = threadIdx.x;
  const float* qr = qc + ((size_t)b * NB + qb) * DM;
  const float* kr = kc + ((size_t)b * NB + j) * DM;
  float s = 0.f;
  for (int d = 0; d < DM; ++d) {
    const float df = qr[d] - kr[d];
    s += df * df;   // sqrt is monotone -> squared distance preserves order
  }
  dist[j] = s;
  __syncthreads();
  if (j == 0) {
    for (int t = 0; t < TOPK; ++t) {
      float best = 3.4e38f; int bi = 0;
      for (int c = 0; c < NB; ++c)
        if (dist[c] < best) { best = dist[c]; bi = c; }
      bidx[((size_t)b * NB + qb) * TOPK + t] = bi;
      dist[bi] = 3.4e38f;
    }
  }
}

// ---------------------------------------------------------------------------
// Kernel 3: block-sparse attention with online softmax, WMMA f16->f32.
// grid (NB, BATCH), 256 threads (8 waves of 32).
__global__ __launch_bounds__(256, 1)
void bsa_attn_kernel(const float* __restrict__ q, const float* __restrict__ k,
                     const float* __restrict__ v, const int* __restrict__ bidx,
                     float* __restrict__ out) {
  __shared__ _Float16 qs[BS * QS_S];   // Q block, f16 row-major
  __shared__ _Float16 ks[BS * KS_S];   // current K block, f16 row-major
  __shared__ _Float16 vt[DM * VT_S];   // current V block, f16 TRANSPOSED [dim][key]
  __shared__ float    sc[BS * SC_S];   // raw scores 64x64
  __shared__ _Float16 ps[BS * PS_S];   // probabilities 64x64 (f16)
  __shared__ float row_m[BS], row_l[BS], row_a[BS];

  const int qb = blockIdx.x, b = blockIdx.y;
  const int tid  = threadIdx.x;
  const int wave = tid >> 5, lane = tid & 31;
  const int n    = lane & 15;            // N index within a 16x16 C tile
  const int mhi  = (lane >> 4) << 3;     // +8 M offset for lanes 16..31

  // ---- selected block indices (uniform scalar loads) ----
  int kbs[TOPK];
#pragma unroll
  for (int j = 0; j < TOPK; ++j)
    kbs[j] = bidx[((b * NB) + qb) * TOPK + j];

  // ---- stage Q once, init softmax state ----
  const size_t qbase = ((size_t)b * SEQ + (size_t)qb * BS) * DM;
#pragma unroll 4
  for (int i = tid; i < BS * DM; i += 256) {
    const int r = i >> 8, c = i & (DM - 1);
    qs[r * QS_S + c] = (_Float16)q[qbase + (size_t)r * DM + c];
  }
  if (tid < BS) { row_m[tid] = -1e30f; row_l[tid] = 0.f; }

  // prefetch first K/V block toward this WGP while Q staging completes
  {
    const size_t pb = ((size_t)b * SEQ + (size_t)kbs[0] * BS) * DM;
    __builtin_prefetch(&k[pb + (size_t)tid * 64], 0, 3);  // 256 x 256B = 64KB
    __builtin_prefetch(&v[pb + (size_t)tid * 64], 0, 3);
  }

  // persistent output accumulators: wave -> (row-tile = wave>>1, col-half = wave&1)
  v8f acc[8];
#pragma unroll
  for (int t = 0; t < 8; ++t)
#pragma unroll
    for (int g = 0; g < 8; ++g) acc[t][g] = 0.f;

  for (int j = 0; j < TOPK; ++j) {
    __syncthreads();   // protect ks/vt (and ps from prior PV reads)
    const int kb = kbs[j];
    const size_t kbase = ((size_t)b * SEQ + (size_t)kb * BS) * DM;
#pragma unroll 4
    for (int i = tid; i < BS * DM; i += 256) {
      const int r = i >> 8, c = i & (DM - 1);
      const float kvv = k[kbase + (size_t)r * DM + c];
      const float vvv = v[kbase + (size_t)r * DM + c];
      ks[r * KS_S + c] = (_Float16)kvv;
      vt[c * VT_S + r] = (_Float16)vvv;     // transpose for PV B-fragments
    }

    // prefetch next block's K/V lines so they arrive during this block's WMMAs
    if (j + 1 < TOPK) {
      const size_t pb = ((size_t)b * SEQ + (size_t)kbs[j + 1] * BS) * DM;
      __builtin_prefetch(&k[pb + (size_t)tid * 64], 0, 3);
      __builtin_prefetch(&v[pb + (size_t)tid * 64], 0, 3);
    }
    __syncthreads();

    // ---- S = Q * K^T (64x64): 16 tiles, 2 per wave ----
    const int rt_s = wave >> 1;
#pragma unroll
    for (int t2 = 0; t2 < 2; ++t2) {
      const int ct = ((wave & 1) << 1) + t2;
      v8f c8;
#pragma unroll
      for (int g = 0; g < 8; ++g) c8[g] = 0.f;
#pragma unroll
      for (int kk = 0; kk < 8; ++kk) {
        v16h a  = frag_a16(qs + rt_s * 16 * QS_S, QS_S, lane, kk * 32);
        v16h bf = frag_b16(ks + ct * 16 * KS_S, KS_S, lane, kk * 32);
        c8 = __builtin_amdgcn_wmma_f32_16x16x32_f16(false, a, false, bf,
                                                    (short)0, c8, false, false);
      }
#pragma unroll
      for (int g = 0; g < 8; ++g)
        sc[(rt_s * 16 + mhi + g) * SC_S + ct * 16 + n] = c8[g];
    }
    __syncthreads();

    // ---- online softmax: one thread per query row ----
    if (tid < BS) {
      const float mo = row_m[tid];
      float mb = -1e30f;
      for (int jj = 0; jj < BS; ++jj)
        mb = fmaxf(mb, sc[tid * SC_S + jj] * SCALE);
      const float mn = fmaxf(mo, mb);
      const float al = __expf(mo - mn);
      float sum = 0.f;
      for (int jj = 0; jj < BS; ++jj) {
        const float p = __expf(sc[tid * SC_S + jj] * SCALE - mn);
        ps[tid * PS_S + jj] = (_Float16)p;
        sum += p;
      }
      row_m[tid] = mn;
      row_l[tid] = row_l[tid] * al + sum;
      row_a[tid] = al;
    }
    __syncthreads();

    // ---- rescale accumulators, O += P * V ----
    const int rt = wave >> 1, h = wave & 1;
    float al8[8];
#pragma unroll
    for (int g = 0; g < 8; ++g) al8[g] = row_a[rt * 16 + mhi + g];
    v16h pa0 = frag_a16(ps + rt * 16 * PS_S, PS_S, lane, 0);
    v16h pa1 = frag_a16(ps + rt * 16 * PS_S, PS_S, lane, 32);
#pragma unroll
    for (int ct = 0; ct < 8; ++ct) {
      const int cto = (h << 3) + ct;
#pragma unroll
      for (int g = 0; g < 8; ++g) acc[ct][g] *= al8[g];
      v16h b0 = frag_b16(vt + cto * 16 * VT_S, VT_S, lane, 0);
      v16h b1 = frag_b16(vt + cto * 16 * VT_S, VT_S, lane, 32);
      acc[ct] = __builtin_amdgcn_wmma_f32_16x16x32_f16(false, pa0, false, b0,
                                                       (short)0, acc[ct], false, false);
      acc[ct] = __builtin_amdgcn_wmma_f32_16x16x32_f16(false, pa1, false, b1,
                                                       (short)0, acc[ct], false, false);
    }
  }

  __syncthreads();
  // ---- normalize and write out ----
  const int rt = wave >> 1, h = wave & 1;
  float li[8];
#pragma unroll
  for (int g = 0; g < 8; ++g) li[g] = 1.0f / row_l[rt * 16 + mhi + g];
#pragma unroll
  for (int ct = 0; ct < 8; ++ct) {
    const int cto = (h << 3) + ct;
#pragma unroll
    for (int g = 0; g < 8; ++g) {
      const size_t row = (size_t)b * SEQ + (size_t)qb * BS + rt * 16 + mhi + g;
      out[row * DM + cto * 16 + n] = acc[ct][g] * li[g];
    }
  }
}

// ---------------------------------------------------------------------------
extern "C" void kernel_launch(void* const* d_in, const int* in_sizes, int n_in,
                              void* d_out, int out_size, void* d_ws, size_t ws_size,
                              hipStream_t stream) {
  const float* q = (const float*)d_in[0];
  const float* k = (const float*)d_in[1];
  const float* v = (const float*)d_in[2];
  float* out = (float*)d_out;

  // workspace: qc (B*NB*DM f32) | kc (B*NB*DM f32) | bidx (B*NB*TOPK int)
  float* qc = (float*)d_ws;
  float* kc = qc + (size_t)BATCH * NB * DM;
  int* bidx = (int*)(kc + (size_t)BATCH * NB * DM);

  bsa_centroid_kernel<<<dim3(NB, BATCH, 2), 256, 0, stream>>>(q, k, qc, kc);
  bsa_topk_kernel<<<dim3(NB, BATCH), 64, 0, stream>>>(qc, kc, bidx);
  bsa_attn_kernel<<<dim3(NB, BATCH), 256, 0, stream>>>(q, k, v, bidx, out);
}
